// MDN_38062000177830
// MI455X (gfx1250) — matrix operand order repaired and checked by reading, not compiled, not run on hardware
//
#include <hip/hip_runtime.h>
#include <hip/hip_bf16.h>

typedef __attribute__((ext_vector_type(16))) _Float16 v16h;
typedef __attribute__((ext_vector_type(8)))  float    v8f;

union V16 { uint4 q[2]; v16h v; };

#define DEPTH   6
#define BATCH   8
#define CDIM    96
#define SPDIM   384
#define HW      576        // 24*24
#define NPIX    9216       // 96*96
#define ROWS    4608       // BATCH*HW rows per layer
#define KCONV   864        // 9*96, K-order: (ky*3+kx)*96 + cin

// workspace byte offsets (all 256-aligned)
#define OFF_ALN   0ull                 // f16 [D][4608][384]     LN'd activations (row-major K)
#define OFF_LWT   21233664ull          // f16 [D][96][384]       linear weights, N-major (c,k)
#define OFF_WTN   21676032ull          // f16 [D][96][864]       conv weights, cout-major, K=t*96+cin
#define OFF_FEA   22671360ull          // f16 [D][4608][96]      linear output, NHWC
#define OFF_OH    27979776ull          // f16 [D][B][9216][96]   upsampled, NHWC (conv input)

// ---------------- weight conversion f32->f16 with layout transforms
__global__ void mdn_convert_w(const float* __restrict__ lin_w,
                              const float* __restrict__ conv_w,
                              _Float16* __restrict__ lwT,
                              _Float16* __restrict__ wTn) {
    int i = blockIdx.x * 256 + threadIdx.x;
    const int n1 = DEPTH * CDIM * SPDIM;      // lwT elements
    const int n2 = DEPTH * CDIM * KCONV;      // wTn elements
    if (i < n1) {
        int k = i % SPDIM;
        int c = (i / SPDIM) % CDIM;
        int d = i / (SPDIM * CDIM);
        lwT[i] = (_Float16)lin_w[((size_t)d * SPDIM + k) * CDIM + c];
    } else if (i < n1 + n2) {
        int j    = i - n1;
        int K    = j % KCONV;                 // t*96 + cin
        int cout = (j / KCONV) % CDIM;
        int d    = j / (KCONV * CDIM);
        int cin  = K % CDIM;
        int t    = K / CDIM;                  // ky*3 + kx
        wTn[j] = (_Float16)conv_w[(((size_t)d * CDIM + cout) * CDIM + cin) * 9 + t];
    }
}

// ---------------- LayerNorm over 384, per (layer,row); write f16 A matrix
__global__ void mdn_ln(const float* __restrict__ sp,
                       const float* __restrict__ ln_g,
                       const float* __restrict__ ln_b,
                       _Float16* __restrict__ aln) {
    int blk = blockIdx.x;            // d*ROWS + row
    int d   = blk / ROWS;
    int row = blk % ROWS;
    const float* x = sp + (size_t)row * SPDIM;
    int t = threadIdx.x;             // 0..127, 3 elems each
    float v0 = x[t * 3 + 0], v1 = x[t * 3 + 1], v2 = x[t * 3 + 2];
    __shared__ float r1[128], r2[128];
    r1[t] = v0 + v1 + v2;
    r2[t] = v0 * v0 + v1 * v1 + v2 * v2;
    __syncthreads();
    for (int off = 64; off > 0; off >>= 1) {
        if (t < off) { r1[t] += r1[t + off]; r2[t] += r2[t + off]; }
        __syncthreads();
    }
    float mu  = r1[0] * (1.0f / 384.0f);
    float var = r2[0] * (1.0f / 384.0f) - mu * mu;
    float inv = rsqrtf(var + 1e-5f);
    _Float16* o = aln + (size_t)blk * SPDIM;
    const float* g = ln_g + d * SPDIM;
    const float* b = ln_b + d * SPDIM;
#pragma unroll
    for (int j = 0; j < 3; ++j) {
        int k = t * 3 + j;
        o[k] = (_Float16)((x[k] - mu) * inv * g[k] + b[k]);
    }
}

// ---------------- Linear 384->96 via WMMA; one 16x16 tile per wave
// A/B fragments are 2x b128 contiguous loads each (ISA 7.12.2 f16 layout).
__global__ void __launch_bounds__(256) mdn_linear(const _Float16* __restrict__ aln,
                                                  const _Float16* __restrict__ lwT,
                                                  const float* __restrict__ lin_b,
                                                  _Float16* __restrict__ fea) {
    int wave = threadIdx.x >> 5;
    int lane = threadIdx.x & 31;
    int tile = blockIdx.x * 8 + wave;           // D * 288 * 6 tiles total
    int d  = tile / (288 * 6);
    int r  = tile % (288 * 6);
    int mt = r / 6, nt = r % 6;
    int m       = lane & 15;
    int laneoff = (lane & 16) ? 8 : 0;          // A K-offset / C M-offset
    int koffB   = (lane & 16) ? 8 : 0;          // B K-offset
    const _Float16* A  = aln + (size_t)d * ROWS * SPDIM + (size_t)(mt * 16 + m) * SPDIM + laneoff;
    const _Float16* Bt = lwT + ((size_t)d * CDIM + nt * 16 + m) * SPDIM + koffB;
    v8f acc = {};
#pragma unroll
    for (int kb = 0; kb < 12; ++kb) {
        int kbase = kb * 32;
        V16 ua, ub;
        ua.q[0] = *(const uint4*)(A + kbase);        // halves 0..7  -> K+0..7
        ua.q[1] = *(const uint4*)(A + kbase + 16);   // halves 8..15 -> K+16..23
        ub.q[0] = *(const uint4*)(Bt + kbase);       // halves 0..7
        ub.q[1] = *(const uint4*)(Bt + kbase + 16);  // halves 8..15
        acc = __builtin_amdgcn_wmma_f32_16x16x32_f16(false, ua.v, false, ub.v,
                                                     (short)0, acc, false, false);
    }
    int c = nt * 16 + m;
    float bias = lin_b[d * CDIM + c];
#pragma unroll
    for (int rr = 0; rr < 8; ++rr) {
        int row = mt * 16 + rr + laneoff;
        fea[((size_t)d * ROWS + row) * CDIM + c] = (_Float16)(acc[rr] + bias);
    }
}

// ---------------- attention upsample 24x24 -> 96x96, NHWC in/out
__global__ void mdn_upsample(const float* __restrict__ prob,
                             const _Float16* __restrict__ fea,
                             _Float16* __restrict__ oh) {
    int t  = blockIdx.x * 256 + threadIdx.x;    // [db][hp][wp][c], c fastest
    int c  = t % CDIM;
    int wp = (t / CDIM) % 96;
    int hp = (t / (CDIM * 96)) % 96;
    int db = t / (CDIM * NPIX);                 // d*BATCH + b
    int h = hp >> 2, w = wp >> 2;
    const _Float16* f = fea + (size_t)db * HW * CDIM + c;
    float acc = 0.0f;
#pragma unroll
    for (int k = 0; k < 9; ++k) {
        int dy = k / 3 - 1, dx = k % 3 - 1;
        int hh = min(max(h + dy, 0), 23);
        int ww = min(max(w + dx, 0), 23);
        float p = prob[(size_t)(db * 9 + k) * NPIX + hp * 96 + wp];
        acc += p * (float)f[(hh * 24 + ww) * CDIM];
    }
    oh[((size_t)db * NPIX + hp * 96 + wp) * CDIM + c] = (_Float16)acc;
}

// ---------------- Conv3x3 (SAME) + bias + BN, implicit GEMM via WMMA
// block: 8 waves -> 8x16 pixels x 16 cout; K = 864 = 9 taps x 96 cin, 27 steps.
// K-order (tap-major, cin-minor) makes (ky,kx) compile-time per step.
__global__ void __launch_bounds__(256) mdn_conv(const _Float16* __restrict__ oh,
                                                const _Float16* __restrict__ wTn,
                                                const float* __restrict__ conv_b,
                                                const float* __restrict__ bn_g,
                                                const float* __restrict__ bn_b,
                                                const float* __restrict__ bn_m,
                                                const float* __restrict__ bn_v,
                                                float* __restrict__ out) {
    __shared__ _Float16 s_in[180 * CDIM];       // [yl*18+xl][cin]  17280 halves
    __shared__ _Float16 s_w[16 * KCONV];        // [n][K]           13824 halves
    int db = blockIdx.y;                        // d*BATCH + b
    int d  = db >> 3;
    int bx   = blockIdx.x;                      // yblk*36 + xblk*6 + nblk
    int yblk = bx / 36;
    int xblk = (bx / 6) % 6;
    int nblk = bx % 6;
    int ybase = yblk * 8, xbase = xblk * 16;

    // stage NHWC input patch (10x18 pixels x 96 ch) as coalesced uint4 copies
    const _Float16* src = oh + (size_t)db * NPIX * CDIM;
    for (int idx = threadIdx.x; idx < 180 * 12; idx += 256) {
        int pix = idx / 12, qi = idx % 12;      // 12 uint4 per 96-ch pixel
        int yl = pix / 18, xl = pix % 18;
        int gy = ybase - 1 + yl, gx = xbase - 1 + xl;
        uint4 val = make_uint4(0u, 0u, 0u, 0u);
        if (gy >= 0 && gy < 96 && gx >= 0 && gx < 96)
            val = *(const uint4*)(src + ((size_t)gy * 96 + gx) * CDIM + qi * 8);
        *(uint4*)(s_in + pix * CDIM + qi * 8) = val;
    }
    // stage weight tile [16][864], contiguous uint4 copies
    const _Float16* wsrc = wTn + ((size_t)d * CDIM + nblk * 16) * KCONV;
    for (int idx = threadIdx.x; idx < 16 * 108; idx += 256) {
        *(uint4*)(s_w + idx * 8) = *(const uint4*)(wsrc + idx * 8);
    }
    __syncthreads();

    int wave = threadIdx.x >> 5;
    int lane = threadIdx.x & 31;
    int m       = lane & 15;
    int laneoff = (lane & 16) ? 8 : 0;
    int koffB   = (lane & 16) ? 8 : 0;
    const _Float16* bp0 = s_w + m * KCONV + koffB;
    v8f acc = {};
#pragma unroll
    for (int kb = 0; kb < 27; ++kb) {
        const int tap  = kb / 3;                // compile-time after unroll
        const int ky   = tap / 3, kx = tap % 3;
        const int cin0 = (kb % 3) * 32;
        const _Float16* ap = s_in + ((wave + ky) * 18 + (m + kx)) * CDIM + cin0 + laneoff;
        V16 ua, ub;
        ua.q[0] = *(const uint4*)ap;            // cin +0..7
        ua.q[1] = *(const uint4*)(ap + 16);     // cin +16..23
        ub.q[0] = *(const uint4*)(bp0 + kb * 32);
        ub.q[1] = *(const uint4*)(bp0 + kb * 32 + 16);
        acc = __builtin_amdgcn_wmma_f32_16x16x32_f16(false, ua.v, false, ub.v,
                                                     (short)0, acc, false, false);
    }

    int cout = nblk * 16 + m;
    float sc   = bn_g[d * CDIM + cout] * rsqrtf(bn_v[d * CDIM + cout] + 1e-5f);
    float sh   = bn_b[d * CDIM + cout] - bn_m[d * CDIM + cout] * sc;
    float bias = conv_b[d * CDIM + cout];
    int y = ybase + wave;
    float* op = out + ((size_t)db * CDIM + cout) * NPIX + y * 96 + xbase;
#pragma unroll
    for (int rr = 0; rr < 8; ++rr) {
        int px = rr + laneoff;
        op[px] = (acc[rr] + bias) * sc + sh;
    }
}

extern "C" void kernel_launch(void* const* d_in, const int* in_sizes, int n_in,
                              void* d_out, int out_size, void* d_ws, size_t ws_size,
                              hipStream_t stream) {
    (void)in_sizes; (void)n_in; (void)out_size; (void)ws_size;
    const float* sp     = (const float*)d_in[0];
    const float* prob   = (const float*)d_in[1];
    const float* ln_g   = (const float*)d_in[2];
    const float* ln_b   = (const float*)d_in[3];
    const float* lin_w  = (const float*)d_in[4];
    const float* lin_b  = (const float*)d_in[5];
    const float* conv_w = (const float*)d_in[6];
    const float* conv_b = (const float*)d_in[7];
    const float* bn_g   = (const float*)d_in[8];
    const float* bn_b   = (const float*)d_in[9];
    const float* bn_m   = (const float*)d_in[10];
    const float* bn_v   = (const float*)d_in[11];
    float* out = (float*)d_out;
    char*  ws  = (char*)d_ws;

    _Float16* aln = (_Float16*)(ws + OFF_ALN);
    _Float16* lwT = (_Float16*)(ws + OFF_LWT);
    _Float16* wTn = (_Float16*)(ws + OFF_WTN);
    _Float16* fea = (_Float16*)(ws + OFF_FEA);
    _Float16* oh  = (_Float16*)(ws + OFF_OH);

    {   // weight conversion + layout transform
        int total = DEPTH * CDIM * SPDIM + DEPTH * CDIM * KCONV;
        mdn_convert_w<<<(total + 255) / 256, 256, 0, stream>>>(lin_w, conv_w, lwT, wTn);
    }
    // LayerNorm (per layer: gamma/beta are per layer)
    mdn_ln<<<DEPTH * ROWS, 128, 0, stream>>>(sp, ln_g, ln_b, aln);
    // Linear 384->96 via WMMA
    mdn_linear<<<(DEPTH * 288 * 6) / 8, 256, 0, stream>>>(aln, lwT, lin_b, fea);
    // attention upsample, NHWC
    mdn_upsample<<<(DEPTH * BATCH * NPIX * CDIM) / 256, 256, 0, stream>>>(prob, fea, oh);
    // Conv3x3 + BN via implicit-GEMM WMMA
    mdn_conv<<<dim3(12 * 6 * 6, DEPTH * BATCH), 256, 0, stream>>>(
        oh, wTn, conv_b, bn_g, bn_b, bn_m, bn_v, out);
}
